// SupernodePooling_39444979646774
// MI455X (gfx1250) — compile-verified, matrix operand order checked by reading, not run on hardware
//
#include <hip/hip_runtime.h>
#include <hip/hip_bf16.h>
#include <math.h>

// SupernodePooling for MI455X (gfx1250, wave32, WMMA).
// H=256, NDIM=3, K=32, S=1024, N=50000.
//
// Pipeline:
//   pack_w   : W1/W2/Wp f32 row-major -> f16 B-fragment-major (WMMA B layout)
//   knn      : 1 block / supernode, register top-32 + LDS argmin rounds
//   mlp      : 1 wave / 16-row tile; embed->WMMA->gelu->LDS->WMMA->col-sum
//   proj     : feats = [sincos(sup_pos) | mean_msgs] ; WMMA vs Wp ; +bp
//
// Workspace layout (bytes):
//   [0,        131072)  w1h  f16  256x256 frag-major
//   [131072,   262144)  w2h  f16  256x256 frag-major
//   [262144,   524288)  wph  f16  512x256 frag-major
//   [524288,   655360)  nbr  i32  1024x32
//   [655360,  2752512)  partial f32 2048x256 (per-16-row-tile mean partials)

typedef __attribute__((ext_vector_type(16))) _Float16 v16h;
typedef __attribute__((ext_vector_type(8)))  float    v8f;

#define LOG2_1E4 13.287712379549449f  // log2(10000)

// A/B fragment element -> K index map (ISA 7.12.2, 16-bit 16x32 A layout):
// lanes 0-15 hold K in {0..7,16..23}, lanes 16-31 hold {8..15,24..31}.
__device__ __forceinline__ int kidx_map(int lane, int e) {
  return ((lane & 16) >> 1) + ((e >> 3) << 4) + (e & 7);
}

__device__ __forceinline__ v8f wmma_f16(v16h a, v16h b, v8f c) {
  return __builtin_amdgcn_wmma_f32_16x16x32_f16(
      /*neg_a=*/false, a, /*neg_b=*/false, b,
      /*c_mod=*/(short)0, c, /*reuse_a=*/false, /*reuse_b=*/false);
}

// gelu (tanh approx) = x * sigmoid(2k(x + 0.044715 x^3)); branch-free:
// one v_exp_f32 + one v_rcp_f32 instead of libm tanhf's divergent path.
__device__ __forceinline__ float gelu_tanh(float x) {
  float u = 0.7978845608028654f * (x + 0.044715f * x * x * x);
  // sigmoid(2u) = 1/(1+exp(-2u)) = 1/(1+exp2(-2u*log2(e)))
  float e = __builtin_amdgcn_exp2f(-2.8853900817779268f * u);
  return x * __builtin_amdgcn_rcpf(1.0f + e);
}

// ---------------------------------------------------------------- pack W ----
__global__ __launch_bounds__(256) void pack_w_kernel(
    const float* __restrict__ W, _Float16* __restrict__ outw,
    int Ktot, int Ntot) {
  int tid = blockIdx.x * 256 + threadIdx.x;
  if (tid >= Ktot * Ntot) return;
  int frag = tid >> 9;          // 512 halves per 32x16 fragment
  int off  = tid & 511;
  int lane = off >> 4;
  int e    = off & 15;
  int ntk  = Ktot >> 5;         // K tiles of 32
  int nt   = frag / ntk;
  int kt   = frag - nt * ntk;
  int k = kt * 32 + kidx_map(lane, e);
  int n = nt * 16 + (lane & 15);
  outw[tid] = (_Float16)W[k * Ntot + n];
}

// ------------------------------------------------------------------- kNN ----
__global__ __launch_bounds__(256) void knn_kernel(
    const float* __restrict__ pos, const int* __restrict__ sup_idx,
    int* __restrict__ nbr, int N) {
  __shared__ unsigned long long rkey[256];
  int s = blockIdx.x, tid = threadIdx.x;
  int si = sup_idx[s];
  float sx = pos[si * 3 + 0], sy = pos[si * 3 + 1], sz = pos[si * 3 + 2];

  float bd[32]; int bi[32];
#pragma unroll
  for (int j = 0; j < 32; ++j) { bd[j] = 3.4e38f; bi[j] = 0x7fffffff; }

  for (int i = tid; i < N; i += 256) {
    float dx = sx - pos[i * 3 + 0];
    float dy = sy - pos[i * 3 + 1];
    float dz = sz - pos[i * 3 + 2];
    float d2 = fmaf(dx, dx, fmaf(dy, dy, dz * dz));
    if (d2 < bd[31]) {                 // most candidates short-circuit here
      float cd = d2; int ci = i;
#pragma unroll
      for (int j = 0; j < 32; ++j) {   // static-index bubble insert
        if (cd < bd[j]) {
          float td = bd[j]; bd[j] = cd; cd = td;
          int   ti = bi[j]; bi[j] = ci; ci = ti;
        }
      }
    }
  }

  // 32 rounds of block-wide argmin over each thread's current head.
  for (int round = 0; round < 32; ++round) {
    unsigned long long mykey =
        ((unsigned long long)__float_as_uint(bd[0]) << 32) |
        (unsigned long long)(unsigned int)bi[0];
    rkey[tid] = mykey;
    __syncthreads();
    for (int off = 128; off > 0; off >>= 1) {
      if (tid < off) {
        unsigned long long o = rkey[tid + off];
        if (o < rkey[tid]) rkey[tid] = o;
      }
      __syncthreads();
    }
    unsigned long long win = rkey[0];
    __syncthreads();
    if (tid == 0) nbr[s * 32 + round] = (int)(unsigned int)(win & 0xffffffffull);
    if (mykey == win) {                // pop my head (unique (dist,idx) keys)
#pragma unroll
      for (int j = 0; j < 31; ++j) { bd[j] = bd[j + 1]; bi[j] = bi[j + 1]; }
      bd[31] = 3.4e38f; bi[31] = 0x7fffffff;
    }
  }
}

// ------------------------------------------------------------ message MLP ----
// One wave per 16-row tile of msg_in (S*K/16 = 2048 tiles). 256 thr = 8 waves.
__global__ __launch_bounds__(256) void mlp_kernel(
    const float* __restrict__ pos, const int* __restrict__ sup_idx,
    const int* __restrict__ nbr,
    const _Float16* __restrict__ w1h, const float* __restrict__ b1,
    const _Float16* __restrict__ w2h, float* __restrict__ partial) {
  __shared__ _Float16 hbuf[8][16 * 256];   // 64 KB: per-wave gelu(h) staging

  int tid = threadIdx.x;
  int wave = tid >> 5, lane = tid & 31;
  int m = lane & 15;
  bool hi = (lane & 16) != 0;
  int tile = blockIdx.x * 8 + wave;        // 0..2047
  int row = tile * 16 + m;                 // global msg row
  int s = row >> 5, k = row & 31;

  __builtin_prefetch(w1h + (size_t)lane * 16, 0, 3);

  // Relative position features for this row (both lane halves duplicate).
  int ni = nbr[s * 32 + k];
  int si = sup_idx[s];
  float rx = pos[si * 3 + 0] - pos[ni * 3 + 0];
  float ry = pos[si * 3 + 1] - pos[ni * 3 + 1];
  float rz = pos[si * 3 + 2] - pos[ni * 3 + 2];
  float dd = sqrtf(rx * rx + ry * ry + rz * rz);

  // Build the 8 A fragments (msg_in embed, ndim=4: 64 ch/coord, half=32).
  // For every unrolled (kt,e): the coordinate index, sin/cos choice, and
  // both lane-halves' frequencies are compile-time constants (the +8 K
  // offset between halves never crosses a 64-channel or sin/cos boundary),
  // so exp2f constant-folds and each element is cndmask+mul+sin/cos.
  v16h a1[8];
#pragma unroll
  for (int kt = 0; kt < 8; ++kt) {
#pragma unroll
    for (int e = 0; e < 16; ++e) {
      const int K0 = kt * 32 + ((e >> 3) << 4) + (e & 7);  // lanes 0-15 K
      const int c  = K0 >> 6;                              // same both halves
      const int w0 = K0 & 63;                              // lo-half channel
      const int w1 = (K0 + 8) & 63;                        // hi-half channel
      float cv = (c == 0) ? rx : ((c == 1) ? ry : ((c == 2) ? rz : dd));
      const float frlo = exp2f(-(float)(w0 & 31) * (LOG2_1E4 / 32.0f));
      const float frhi = exp2f(-(float)(w1 & 31) * (LOG2_1E4 / 32.0f));
      float ang = cv * (hi ? frhi : frlo);
      float v = (w0 < 32) ? __sinf(ang) : __cosf(ang);     // static choice
      a1[kt][e] = (_Float16)v;
    }
  }

  _Float16* hl = &hbuf[wave][0];
  int mrow = (lane >> 4) << 3;             // 0 or 8 (C-frag M offset)

  // GEMM1: h = gelu(msg_in @ W1 + b1) -> LDS (f16, A-loadable layout)
  for (int nt = 0; nt < 16; ++nt) {
    v8f c = {};
#pragma unroll
    for (int kt = 0; kt < 8; ++kt) {
      v16h b = *(const v16h*)(w1h + (size_t)(nt * 8 + kt) * 512 + lane * 16);
      c = wmma_f16(a1[kt], b, c);
    }
    float bn = b1[nt * 16 + m];
#pragma unroll
    for (int r = 0; r < 8; ++r) {
      float g = gelu_tanh(c[r] + bn);
      hl[(mrow + r) * 256 + nt * 16 + m] = (_Float16)g;
    }
  }
  __syncthreads();

  // Reload h as A fragments (K pairs are contiguous -> b32 LDS loads).
  v16h a2[8];
#pragma unroll
  for (int kt = 0; kt < 8; ++kt) {
#pragma unroll
    for (int p = 0; p < 8; ++p) {
      int K = kt * 32 + kidx_map(lane, 2 * p);
      union { unsigned int u; _Float16 h[2]; } pr;
      pr.u = *(const unsigned int*)(hl + m * 256 + K);
      a2[kt][2 * p]     = pr.h[0];
      a2[kt][2 * p + 1] = pr.h[1];
    }
  }

  // GEMM2: msgs = h @ W2 ; column-sum over this tile's 16 rows (mean/32).
  for (int nt = 0; nt < 16; ++nt) {
    v8f c = {};
#pragma unroll
    for (int kt = 0; kt < 8; ++kt) {
      v16h b = *(const v16h*)(w2h + (size_t)(nt * 8 + kt) * 512 + lane * 16);
      c = wmma_f16(a2[kt], b, c);
    }
    float sum = 0.0f;
#pragma unroll
    for (int r = 0; r < 8; ++r) sum += c[r];
    sum += __shfl_xor(sum, 16, 32);        // add other 8-row half
    if (lane < 16)
      partial[(size_t)tile * 256 + nt * 16 + m] = sum * (1.0f / 32.0f);
  }
}

// ------------------------------------------------------- final projection ----
// One block per 16 supernodes; feats = [sincos(sup_pos) | agg], out = feats@Wp+bp
__global__ __launch_bounds__(256) void proj_kernel(
    const float* __restrict__ pos, const int* __restrict__ sup_idx,
    const float* __restrict__ partial, const float* __restrict__ b2,
    const _Float16* __restrict__ wph, const float* __restrict__ bp,
    float* __restrict__ out) {
  __shared__ _Float16 feats[16 * 512];     // 16 KB
  int tid = threadIdx.x;
  int s0 = blockIdx.x * 16;

  for (int i = tid; i < 16 * 512; i += 256) {
    int mm = i >> 9;
    int j  = i & 511;
    int s  = s0 + mm;
    float v;
    if (j < 256) {
      // sup_emb: ndim=3, dpn=85, half=42 -> 84 ch/coord, 4 zero-pad channels.
      if (j < 252) {
        int c = j / 84, w = j % 84;
        int si = sup_idx[s];
        float p = pos[si * 3 + c];
        float fr = __builtin_amdgcn_exp2f(-(float)(w % 42) *
                                          (LOG2_1E4 / 42.0f));
        float ang = p * fr;
        v = (w < 42) ? __sinf(ang) : __cosf(ang);
      } else {
        v = 0.0f;
      }
    } else {
      int jj = j - 256;                    // agg = mean over both row-tiles + b2
      v = partial[(size_t)(2 * s) * 256 + jj] +
          partial[(size_t)(2 * s + 1) * 256 + jj] + b2[jj];
    }
    feats[i] = (_Float16)v;
  }
  __syncthreads();

  int wave = tid >> 5, lane = tid & 31;
  int m = lane & 15;
  int mrow = (lane >> 4) << 3;

  // A fragments for this 16x512 tile (shared across both n-tiles of the wave).
  v16h af[16];
#pragma unroll
  for (int kt = 0; kt < 16; ++kt) {
#pragma unroll
    for (int p = 0; p < 8; ++p) {
      int K = kt * 32 + kidx_map(lane, 2 * p);
      union { unsigned int u; _Float16 h[2]; } pr;
      pr.u = *(const unsigned int*)(&feats[m * 512 + K]);
      af[kt][2 * p]     = pr.h[0];
      af[kt][2 * p + 1] = pr.h[1];
    }
  }

  for (int t = 0; t < 2; ++t) {
    int nt = wave * 2 + t;                 // 16 n-tiles over 8 waves
    v8f c = {};
#pragma unroll
    for (int kt = 0; kt < 16; ++kt) {
      v16h b = *(const v16h*)(wph + (size_t)(nt * 16 + kt) * 512 + lane * 16);
      c = wmma_f16(af[kt], b, c);
    }
    float bpv = bp[nt * 16 + m];
#pragma unroll
    for (int r = 0; r < 8; ++r) {
      out[(size_t)(s0 + mrow + r) * 256 + nt * 16 + m] = c[r] + bpv;
    }
  }
}

// ----------------------------------------------------------------- launch ----
extern "C" void kernel_launch(void* const* d_in, const int* in_sizes, int n_in,
                              void* d_out, int out_size, void* d_ws,
                              size_t ws_size, hipStream_t stream) {
  const float* positions = (const float*)d_in[0];
  const int*   sup_idx   = (const int*)d_in[1];
  const float* W1 = (const float*)d_in[2];
  const float* b1 = (const float*)d_in[3];
  const float* W2 = (const float*)d_in[4];
  const float* b2 = (const float*)d_in[5];
  const float* Wp = (const float*)d_in[6];
  const float* bp = (const float*)d_in[7];
  float* out = (float*)d_out;

  int N = in_sizes[0] / 3;   // 50000
  int S = in_sizes[1];       // 1024

  char* ws = (char*)d_ws;
  _Float16* w1h     = (_Float16*)(ws);
  _Float16* w2h     = (_Float16*)(ws + 131072);
  _Float16* wph     = (_Float16*)(ws + 262144);
  int*      nbr     = (int*)(ws + 524288);
  float*    partial = (float*)(ws + 655360);

  pack_w_kernel<<<(256 * 256) / 256, 256, 0, stream>>>(W1, w1h, 256, 256);
  pack_w_kernel<<<(256 * 256) / 256, 256, 0, stream>>>(W2, w2h, 256, 256);
  pack_w_kernel<<<(512 * 256) / 256, 256, 0, stream>>>(Wp, wph, 512, 256);

  knn_kernel<<<S, 256, 0, stream>>>(positions, sup_idx, nbr, N);

  // S*K rows / 16 rows-per-wave / 8 waves-per-block = 256 blocks
  mlp_kernel<<<(S * 32) / (16 * 8), 256, 0, stream>>>(
      positions, sup_idx, nbr, w1h, b1, w2h, partial);

  proj_kernel<<<S / 16, 256, 0, stream>>>(
      positions, sup_idx, partial, b2, wph, bp, out);
}
